// CrossAttention_3942779977831
// MI455X (gfx1250) — matrix operand order repaired and checked by reading, not compiled, not run on hardware
//
#include <hip/hip_runtime.h>
#include <hip/hip_bf16.h>
#include <cstdint>

// ---------------------------------------------------------------------------
// CrossAttention (self-attn): B=2, N=4096, D=512, H=8, DH=64
// bf16 WMMA (v_wmma_f32_16x16x32_bf16) for all GEMMs, fp32 online softmax,
// double-buffered K/V staging with async global->LDS copies when available.
// ---------------------------------------------------------------------------

#define Bsz  2
#define Nseq 4096
#define Dm   512
#define Hh   8
#define DHd  64
#define SCALE 0.125f  // 64^-0.5

typedef __attribute__((ext_vector_type(16))) __bf16 v16bf;
typedef __attribute__((ext_vector_type(8)))  float  v8f;

union Frag16 {
    uint4 q[2];
    v16bf v;
};

// native f32 -> bf16 (RNE fptrunc; backend picks the cvt/pack instructions)
__device__ __forceinline__ unsigned short f2bf(float f) {
    __bf16 h = (__bf16)f;
    return __builtin_bit_cast(unsigned short, h);
}
__device__ __forceinline__ unsigned int f2bf2(float lo, float hi) {
    return (unsigned int)f2bf(lo) | ((unsigned int)f2bf(hi) << 16);
}

__device__ __forceinline__ v8f wmma_bf16(v16bf a, v16bf b, v8f c) {
    // (neg_a, A, neg_b, B, c_mod, C, reuse_a, reuse_b)
    return __builtin_amdgcn_wmma_f32_16x16x32_bf16(false, a, false, b,
                                                   (short)0, c, false, false);
}

#if defined(__gfx1250__) && __has_builtin(__builtin_amdgcn_global_load_async_to_lds_b128)
#define HAVE_ASYNC_LDS 1
typedef int int4v __attribute__((vector_size(4 * sizeof(int))));
typedef __attribute__((address_space(1))) int4v* gas_i4p;   // global int4*
typedef __attribute__((address_space(3))) int4v* las_i4p;   // LDS int4*
#else
#define HAVE_ASYNC_LDS 0
#endif

__device__ __forceinline__ void wait_async_le(int n_is_two) {
#if HAVE_ASYNC_LDS
#if __has_builtin(__builtin_amdgcn_s_wait_asynccnt)
    if (n_is_two) __builtin_amdgcn_s_wait_asynccnt(2);
    else          __builtin_amdgcn_s_wait_asynccnt(0);
#else
    if (n_is_two) asm volatile("s_wait_asynccnt 0x2" ::: "memory");
    else          asm volatile("s_wait_asynccnt 0x0" ::: "memory");
#endif
#else
    (void)n_is_two;
#endif
}

// ---------------------------------------------------------------------------
// Kernel 1: QKV projection.  C[m0:m0+64][n0:n0+64] = x @ W{q,k,v}, bf16 out
// in head-split layout [B][H][N][DH].  SCALE is folded into Q here.
// Block = 128 threads = 4 waves; each wave computes a 16x64 strip.
// ---------------------------------------------------------------------------
#define GSTR 40   // padded LDS row stride (ushorts): 80B -> conflict-free

__global__ __launch_bounds__(128) void qkv_gemm_kernel(
    const float* __restrict__ x,
    const float* __restrict__ Wq, const float* __restrict__ Wk,
    const float* __restrict__ Wv,
    unsigned short* __restrict__ q_ws, unsigned short* __restrict__ k_ws,
    unsigned short* __restrict__ v_ws)
{
    __shared__ __align__(32) unsigned short sA [64 * GSTR];  // x tile bf16
    __shared__ __align__(32) unsigned short sBt[64 * GSTR];  // W^T tile [n][k]

    const int m0 = blockIdx.x * 64;
    const int n0 = blockIdx.y * 64;
    const int which = blockIdx.z;
    const float* W = (which == 0) ? Wq : (which == 1) ? Wk : Wv;
    unsigned short* dst = (which == 0) ? q_ws : (which == 1) ? k_ws : v_ws;
    const float oscale = (which == 0) ? SCALE : 1.0f;

    const int tid  = threadIdx.x;
    const int wave = tid >> 5;
    const int lane = tid & 31;
    const int half = lane >> 4;
    const int l16  = lane & 15;

    v8f acc[4] = {};

    for (int k0 = 0; k0 < Dm; k0 += 32) {
        __syncthreads();
        // ---- stage A: x[m0..m0+63][k0..k0+31] -> bf16 sA (padded rows)
        {
            const int base = tid * 16;
            const int row  = base >> 5;
            const int col  = base & 31;
            const float* src = x + (size_t)(m0 + row) * Dm + k0 + col;
            #pragma unroll
            for (int c = 0; c < 4; ++c) {
                float4 f4 = *(const float4*)(src + 4 * c);
                const int o = row * GSTR + col + 4 * c;
                *(unsigned int*)&sA[o]     = f2bf2(f4.x, f4.y);
                *(unsigned int*)&sA[o + 2] = f2bf2(f4.z, f4.w);
            }
        }
        // ---- stage Bt: W[k0+k][n0+n] -> sBt[n][k]
        {
            const int base = tid * 16;
            const int k = base >> 6;
            const int n = base & 63;
            const float* src = W + (size_t)(k0 + k) * Dm + n0 + n;
            #pragma unroll
            for (int c = 0; c < 4; ++c) {
                float4 f4 = *(const float4*)(src + 4 * c);
                const int nn = n + 4 * c;
                sBt[(nn + 0) * GSTR + k] = f2bf(f4.x);
                sBt[(nn + 1) * GSTR + k] = f2bf(f4.y);
                sBt[(nn + 2) * GSTR + k] = f2bf(f4.z);
                sBt[(nn + 3) * GSTR + k] = f2bf(f4.w);
            }
        }
        __syncthreads();

        // ---- A fragment (contiguous k = 16*half + s, matches B)
        Frag16 a;
        {
            const unsigned short* p = &sA[(wave * 16 + l16) * GSTR];
            a.q[0] = *(const uint4*)(p + half * 16);
            a.q[1] = *(const uint4*)(p + half * 16 + 8);
        }
        #pragma unroll
        for (int f = 0; f < 4; ++f) {
            Frag16 bfr;
            const unsigned short* p = &sBt[(f * 16 + l16) * GSTR + half * 16];
            bfr.q[0] = *(const uint4*)(p);
            bfr.q[1] = *(const uint4*)(p + 8);
            acc[f] = wmma_bf16(a.v, bfr.v, acc[f]);
        }
    }

    // ---- epilogue: scatter bf16 into head-split [b][h][n][dh]
    #pragma unroll
    for (int f = 0; f < 4; ++f) {
        const int c  = n0 + f * 16 + l16;
        const int h  = c >> 6;
        const int dh = c & 63;
        #pragma unroll
        for (int r = 0; r < 8; ++r) {
            const int i = m0 + wave * 16 + (half ? 8 + r : r);
            const int b = i >> 12;
            const int n = i & 4095;
            dst[(((size_t)(b * Hh + h) * Nseq + n) * DHd) + dh] =
                f2bf(acc[f][r] * oscale);
        }
    }
}

// ---------------------------------------------------------------------------
// Kernel 2: flash attention.  grid = (N/64, H, B), block = 128 (4 waves).
// Double-buffered 32-wide KV tiles; K staged via async global->LDS copies.
// ---------------------------------------------------------------------------
#define KSTR 72   // sK row stride  (144B -> 16 distinct banks, 16B aligned)
#define VSTR 40   // sVT row stride (80B  -> 16 distinct banks, 16B aligned)
#define PSTR 40   // sP  row stride

__global__ __launch_bounds__(128) void attn_kernel(
    const unsigned short* __restrict__ q_ws,
    const unsigned short* __restrict__ k_ws,
    const unsigned short* __restrict__ v_ws,
    unsigned short* __restrict__ o_ws)   // merged heads: [B][N][H*DH] bf16
{
    __shared__ __align__(32) unsigned short sKb [2][32 * KSTR];  // K  [j][d]
    __shared__ __align__(32) unsigned short sVTb[2][64 * VSTR];  // V^T [d][j]
    __shared__ __align__(32) unsigned short sPb [4][16 * PSTR];  // per-wave P

    const int it0 = blockIdx.x * 64;
    const int h   = blockIdx.y;
    const int b   = blockIdx.z;
    const int tid  = threadIdx.x;
    const int wave = tid >> 5;
    const int lane = tid & 31;
    const int half = lane >> 4;
    const int l16  = lane & 15;

    const size_t headoff = (size_t)(b * Hh + h) * Nseq * DHd;
    const unsigned short* Q = q_ws + headoff;
    const unsigned short* K = k_ws + headoff;
    const unsigned short* V = v_ws + headoff;

    // staging coordinates (each thread moves 16 ushorts = 32B per tile)
    const int base = tid * 16;
    const int srow = base >> 6;       // kv row j   (0..31)
    const int scol = base & 63;       // dim d col  {0,16,32,48}

    // ---- preload Q A-fragments (Q already carries SCALE)
    Frag16 aQ[2];
    {
        const unsigned short* qrow = Q + (size_t)(it0 + wave * 16 + l16) * DHd;
        #pragma unroll
        for (int s = 0; s < 2; ++s) {
            aQ[s].q[0] = *(const uint4*)(qrow + 32 * s + half * 16);
            aQ[s].q[1] = *(const uint4*)(qrow + 32 * s + half * 16 + 8);
        }
    }

    v8f o[4] = {};
    float m_run[8], l_run[8];
    #pragma unroll
    for (int r = 0; r < 8; ++r) { m_run[r] = -1e30f; l_run[r] = 0.0f; }

    // ---- prologue: stage tile 0 (K -> LDS buf0, V -> registers)
    __align__(16) unsigned short vtmp[16];
    {
        const unsigned short* gk = K + (size_t)srow * DHd + scol;
#if HAVE_ASYNC_LDS
        gas_i4p g = (gas_i4p)(uintptr_t)gk;
        las_i4p l = (las_i4p)(unsigned int)(uintptr_t)&sKb[0][srow * KSTR + scol];
        __builtin_amdgcn_global_load_async_to_lds_b128(g, l, 0, 0);
        __builtin_amdgcn_global_load_async_to_lds_b128(g, l, 16, 0);
#else
        *(uint4*)&sKb[0][srow * KSTR + scol]     = *(const uint4*)(gk);
        *(uint4*)&sKb[0][srow * KSTR + scol + 8] = *(const uint4*)(gk + 8);
#endif
        const unsigned short* gv = V + (size_t)srow * DHd + scol;
        *(uint4*)&vtmp[0] = *(const uint4*)(gv);
        *(uint4*)&vtmp[8] = *(const uint4*)(gv + 8);
    }

    const int ntiles = Nseq / 32;   // 128
    int curb = 0;
    for (int it = 0; it < ntiles; ++it, curb ^= 1) {
        // ---- scatter this tile's V (in regs) into transposed LDS buffer
        #pragma unroll
        for (int e = 0; e < 16; ++e)
            sVTb[curb][(scol + e) * VSTR + srow] = vtmp[e];

        // ---- prefetch next tile (K async, V to registers)
        const int haveNext = (it + 1) < ntiles;
        if (haveNext) {
            const int jt2 = (it + 1) * 32;
            const unsigned short* gk = K + (size_t)(jt2 + srow) * DHd + scol;
#if HAVE_ASYNC_LDS
            gas_i4p g = (gas_i4p)(uintptr_t)gk;
            las_i4p l =
                (las_i4p)(unsigned int)(uintptr_t)&sKb[curb ^ 1][srow * KSTR + scol];
            __builtin_amdgcn_global_load_async_to_lds_b128(g, l, 0, 0);
            __builtin_amdgcn_global_load_async_to_lds_b128(g, l, 16, 0);
#else
            *(uint4*)&sKb[curb ^ 1][srow * KSTR + scol]     = *(const uint4*)(gk);
            *(uint4*)&sKb[curb ^ 1][srow * KSTR + scol + 8] = *(const uint4*)(gk + 8);
#endif
            const unsigned short* gv = V + (size_t)(jt2 + srow) * DHd + scol;
            *(uint4*)&vtmp[0] = *(const uint4*)(gv);
            *(uint4*)&vtmp[8] = *(const uint4*)(gv + 8);
        }
        wait_async_le(haveNext);   // current K tile complete (async in-order)
        __syncthreads();           // K/V tile visible to all waves

        // ---- S = Q K^T  (16x32 fp32, two column tiles, Kdim=64 -> 2 steps)
        v8f sacc[2] = {};
        #pragma unroll
        for (int ct = 0; ct < 2; ++ct) {
            #pragma unroll
            for (int s = 0; s < 2; ++s) {
                Frag16 bK;
                const unsigned short* p =
                    &sKb[curb][(ct * 16 + l16) * KSTR + 32 * s + half * 16];
                bK.q[0] = *(const uint4*)(p);
                bK.q[1] = *(const uint4*)(p + 8);
                sacc[ct] = wmma_bf16(aQ[s].v, bK.v, sacc[ct]);
            }
        }

        // ---- online softmax (per register row; 16-lane butterflies)
        unsigned short* sP = &sPb[wave][0];
        #pragma unroll
        for (int r = 0; r < 8; ++r) {
            const float s0 = sacc[0][r];
            const float s1 = sacc[1][r];
            float mx = fmaxf(s0, s1);
            #pragma unroll
            for (int off = 1; off < 16; off <<= 1)
                mx = fmaxf(mx, __shfl_xor(mx, off, 32));
            const float newm = fmaxf(m_run[r], mx);
            const float corr = __expf(m_run[r] - newm);
            const float p0 = __expf(s0 - newm);
            const float p1 = __expf(s1 - newm);
            float rs = p0 + p1;
            #pragma unroll
            for (int off = 1; off < 16; off <<= 1)
                rs += __shfl_xor(rs, off, 32);
            l_run[r] = l_run[r] * corr + rs;
            m_run[r] = newm;
            #pragma unroll
            for (int f = 0; f < 4; ++f) o[f][r] *= corr;
            const int prow = half ? 8 + r : r;
            sP[prow * PSTR + l16]      = f2bf(p0);
            sP[prow * PSTR + 16 + l16] = f2bf(p1);
        }
        // wave-local LDS ordering (DS ops complete in order within a wave)

        // ---- O += P V
        Frag16 aP;
        {
            const unsigned short* p = &sP[l16 * PSTR];
            aP.q[0] = *(const uint4*)(p + half * 16);
            aP.q[1] = *(const uint4*)(p + half * 16 + 8);
        }
        #pragma unroll
        for (int f = 0; f < 4; ++f) {
            Frag16 bV;
            const unsigned short* p =
                &sVTb[curb][(f * 16 + l16) * VSTR + half * 16];
            bV.q[0] = *(const uint4*)(p);
            bV.q[1] = *(const uint4*)(p + 8);
            o[f] = wmma_bf16(aP.v, bV.v, o[f]);
        }
        __syncthreads();  // all reads of buf[curb] done before it is restaged
    }

    // ---- normalize and store merged-head bf16 [b][n][h*64+d]
    #pragma unroll
    for (int r = 0; r < 8; ++r) {
        const float inv = 1.0f / l_run[r];
        const int i = it0 + wave * 16 + (half ? 8 + r : r);
        const size_t rowoff = ((size_t)b * Nseq + i) * (Hh * DHd) + h * DHd;
        #pragma unroll
        for (int f = 0; f < 4; ++f)
            o_ws[rowoff + f * 16 + l16] = f2bf(o[f][r] * inv);
    }
}

// ---------------------------------------------------------------------------
// Kernel 3: output projection.  out = O @ Wo + bo  (fp32 out)
// ---------------------------------------------------------------------------
__global__ __launch_bounds__(128) void out_gemm_kernel(
    const unsigned short* __restrict__ o_ws,   // [B*N][512] bf16
    const float* __restrict__ Wo, const float* __restrict__ bo,
    float* __restrict__ out)
{
    __shared__ __align__(32) unsigned short sBt[64 * GSTR];  // Wo^T [n][k]

    const int m0 = blockIdx.x * 64;
    const int n0 = blockIdx.y * 64;
    const int tid  = threadIdx.x;
    const int wave = tid >> 5;
    const int lane = tid & 31;
    const int half = lane >> 4;
    const int l16  = lane & 15;

    v8f acc[4] = {};

    for (int k0 = 0; k0 < Dm; k0 += 32) {
        __syncthreads();
        {
            const int base = tid * 16;
            const int k = base >> 6;
            const int n = base & 63;
            const float* src = Wo + (size_t)(k0 + k) * Dm + n0 + n;
            #pragma unroll
            for (int c = 0; c < 4; ++c) {
                float4 f4 = *(const float4*)(src + 4 * c);
                const int nn = n + 4 * c;
                sBt[(nn + 0) * GSTR + k] = f2bf(f4.x);
                sBt[(nn + 1) * GSTR + k] = f2bf(f4.y);
                sBt[(nn + 2) * GSTR + k] = f2bf(f4.z);
                sBt[(nn + 3) * GSTR + k] = f2bf(f4.w);
            }
        }
        __syncthreads();

        Frag16 a;
        {
            const unsigned short* p =
                o_ws + (size_t)(m0 + wave * 16 + l16) * Dm + k0;
            a.q[0] = *(const uint4*)(p + half * 16);
            a.q[1] = *(const uint4*)(p + half * 16 + 8);
        }
        #pragma unroll
        for (int f = 0; f < 4; ++f) {
            Frag16 bfr;
            const unsigned short* p = &sBt[(f * 16 + l16) * GSTR + half * 16];
            bfr.q[0] = *(const uint4*)(p);
            bfr.q[1] = *(const uint4*)(p + 8);
            acc[f] = wmma_bf16(a.v, bfr.v, acc[f]);
        }
    }

    #pragma unroll
    for (int f = 0; f < 4; ++f) {
        const int c = n0 + f * 16 + l16;
        const float bias = bo[c];
        #pragma unroll
        for (int r = 0; r < 8; ++r) {
            const int i = m0 + wave * 16 + (half ? 8 + r : r);
            out[(size_t)i * Dm + c] = acc[f][r] + bias;
        }
    }
}

// ---------------------------------------------------------------------------
extern "C" void kernel_launch(void* const* d_in, const int* in_sizes, int n_in,
                              void* d_out, int out_size, void* d_ws, size_t ws_size,
                              hipStream_t stream)
{
    (void)in_sizes; (void)n_in; (void)out_size; (void)ws_size;

    const float* x  = (const float*)d_in[0];
    const float* Wq = (const float*)d_in[1];
    const float* Wk = (const float*)d_in[2];
    const float* Wv = (const float*)d_in[3];
    const float* Wo = (const float*)d_in[4];
    const float* bo = (const float*)d_in[5];
    float* out = (float*)d_out;

    const size_t elems = (size_t)Bsz * Hh * Nseq * DHd;   // 4,194,304
    unsigned short* q_ws = (unsigned short*)d_ws;
    unsigned short* k_ws = q_ws + elems;
    unsigned short* v_ws = k_ws + elems;
    unsigned short* o_ws = v_ws + elems;                  // [B*N][512] bf16

    dim3 g1((Bsz * Nseq) / 64, Dm / 64, 3);
    qkv_gemm_kernel<<<g1, 128, 0, stream>>>(x, Wq, Wk, Wv, q_ws, k_ws, v_ws);

    dim3 g2(Nseq / 64, Hh, Bsz);
    attn_kernel<<<g2, 128, 0, stream>>>(q_ws, k_ws, v_ws, o_ws);

    dim3 g3((Bsz * Nseq) / 64, Dm / 64);
    out_gemm_kernel<<<g3, 128, 0, stream>>>(o_ws, Wo, bo, out);
}
